// EquivariantAttention_30580167147934
// MI455X (gfx1250) — matrix operand order repaired and checked by reading, not compiled, not run on hardware
//
#include <hip/hip_runtime.h>
#include <hip/hip_bf16.h>

typedef _Float16 v16h __attribute__((ext_vector_type(16)));
typedef _Float16 v8h  __attribute__((ext_vector_type(8)));
typedef _Float16 v4h  __attribute__((ext_vector_type(4)));
typedef _Float16 v2h  __attribute__((ext_vector_type(2)));
typedef float    v8f  __attribute__((ext_vector_type(8)));

constexpr int kB  = 4;
constexpr int kS  = 512;
constexpr int kD  = 512;
constexpr int kH  = 8;
constexpr int kDH = 64;

__device__ inline v16h concat8(v8h lo, v8h hi) {
    return __builtin_shufflevector(lo, hi, 0,1,2,3,4,5,6,7,8,9,10,11,12,13,14,15);
}

// ---------------------------------------------------------------------------
// Fragment loaders (wave32, f16). All loads are contiguous 16B vectors.
//
// A 16x32 (row-major source, rows = M, cols = K):
//   lane<16:  M=lane,    K in {k0+0..7,  k0+16..23}
//   lane>=16: M=lane-16, K in {k0+8..15, k0+24..31}
// ---------------------------------------------------------------------------
__device__ inline v16h load_a_frag_rm(const _Float16* base, int ld, int row0, int k0, int lane) {
    int hid = lane >> 4;
    int m   = lane & 15;
    const _Float16* p = base + (size_t)(row0 + m) * ld + k0 + hid * 8;
    v8h lo = *(const v8h*)p;
    v8h hi = *(const v8h*)(p + 16);
    return concat8(lo, hi);
}

// B 32x16 fragment from a K-transposed tile (rows = N, cols = K):
//   lane's column n = lane&15 needs K in {k0+hid*8+0..7, k0+16+hid*8+0..7} (matches A).
__device__ inline v16h load_b_frag_kt(const _Float16* base, int ld, int k0, int n0, int lane) {
    int hid = lane >> 4;
    int n   = lane & 15;
    const _Float16* p = base + (size_t)(n0 + n) * ld + k0 + hid * 8;
    v8h lo = *(const v8h*)p;
    v8h hi = *(const v8h*)(p + 16);
    return concat8(lo, hi);
}

// ---------------------------------------------------------------------------
// Kernel 1: pos_w[i,j,h] = PE[i,j,:] @ W_pos[:,h] + b_pos[h]  (pair-major out)
// The roofline limiter: streams 512MB of PE once (~22us @ 23.3 TB/s).
// Done as a WMMA GEMM (M = S*S pairs, K = 512, N = 8 padded to 16) so that the
// per-byte instruction-issue cost stays far below the HBM rate: ~25 issue
// slots per 2KB row instead of ~170 for a VALU dot-product version.
// 128 threads = 4 waves; each block covers 64 pairs; K staged in 128 chunks.
// ---------------------------------------------------------------------------
__global__ __launch_bounds__(128) void pos_bias_wmma(
    const float* __restrict__ pe,     // (S,S,D)
    const float* __restrict__ w_pos,  // (D,H)
    const float* __restrict__ b_pos,  // (H)
    float* __restrict__ pos_w)        // (S,S,H)
{
    __shared__ _Float16 As[64][136];  // 64 pairs x 128-k chunk (17KB)
    __shared__ _Float16 Bt[16][520];  // whole W_pos, K-transposed, cols 8..15 = 0

    int tid = threadIdx.x;
    int wave = tid >> 5, lane = tid & 31;
    int hid = lane >> 4, l16 = lane & 15;
    size_t pair0 = (size_t)blockIdx.x * 64;

    // Stage all of W_pos once (f32 -> f16, K-transposed, zero-padded N)
    for (int idx = tid; idx < 16 * 512; idx += 128) {
        int n = idx & 15, k = idx >> 4;
        Bt[n][k] = (n < kH) ? (_Float16)w_pos[(size_t)k * kH + n] : (_Float16)0.0f;
    }

    v8f acc = {};
    for (int k0 = 0; k0 < kD; k0 += 128) {
        __syncthreads();
        // Stage 64x128 PE chunk: coalesced float4 reads, packed v4h stores
        for (int idx = tid; idx < 64 * 32; idx += 128) {
            int r = idx >> 5, c4 = (idx & 31) * 4;
            float4 v = *(const float4*)(pe + (pair0 + r) * kD + k0 + c4);
            if (k0 < kD - 128)  // stream-ahead hint for the next chunk
                __builtin_prefetch(pe + (pair0 + r) * kD + k0 + 128 + c4, 0, 1);
            v4h pk = { (_Float16)v.x, (_Float16)v.y, (_Float16)v.z, (_Float16)v.w };
            *(v4h*)&As[r][c4] = pk;
        }
        __syncthreads();
#pragma unroll
        for (int kk = 0; kk < 128; kk += 32) {
            v16h af = load_a_frag_rm(&As[0][0], 136, wave * 16, kk, lane);
            v16h bf = load_b_frag_kt(&Bt[0][0], 520, k0 + kk, 0, lane);
            acc = __builtin_amdgcn_wmma_f32_16x16x32_f16(
                false, af, false, bf, (short)0, acc, false, false);
        }
    }

    // C layout: row (pair) = wave*16 + r + hid*8, col (h) = l16.
    // Lanes 0..7 write 8 consecutive floats -> 32B contiguous stores per row.
    float bias = (l16 < kH) ? b_pos[l16] : 0.0f;
#pragma unroll
    for (int r = 0; r < 8; ++r) {
        if (l16 < kH) {
            size_t pair = pair0 + wave * 16 + r + hid * 8;
            pos_w[pair * kH + l16] = acc[r] + bias;
        }
    }
}

// ---------------------------------------------------------------------------
// Kernel 2/4: C[M,N] = A[M,K] @ Bm[K,N] + bias via f16 WMMA.
// A tile row-major, B tile stored K-transposed so every fragment load is b128.
// 128 threads = 4 waves; 64x64 tile; K-step 32; wave owns 16 rows x 64 cols.
// ---------------------------------------------------------------------------
template <typename AT, typename OT>
__global__ __launch_bounds__(128) void gemm_bias_wmma(
    const AT* __restrict__ A, const float* __restrict__ Bm,
    const float* __restrict__ bias, OT* __restrict__ C,
    int M, int N, int K)
{
    __shared__ _Float16 As[64][40];   // [m][k]
    __shared__ _Float16 Bt[64][40];   // [n][k] (K-transposed)

    int tid = threadIdx.x;
    int wave = tid >> 5, lane = tid & 31;
    int hid = lane >> 4, l16 = lane & 15;
    int m0 = blockIdx.y * 64;
    int n0 = blockIdx.x * 64;

    v8f acc[4] = {};

    for (int k0 = 0; k0 < K; k0 += 32) {
        __syncthreads();
        for (int idx = tid; idx < 64 * 16; idx += 128) {
            int r = idx >> 4, c2 = (idx & 15) * 2;
            v2h pk = { (_Float16)A[(size_t)(m0 + r) * K + k0 + c2],
                       (_Float16)A[(size_t)(m0 + r) * K + k0 + c2 + 1] };
            *(v2h*)&As[r][c2] = pk;
        }
        for (int idx = tid; idx < 64 * 16; idx += 128) {
            int n  = idx & 63;
            int kk = (idx >> 6) * 2;
            v2h pk = { (_Float16)Bm[(size_t)(k0 + kk) * N + n0 + n],
                       (_Float16)Bm[(size_t)(k0 + kk + 1) * N + n0 + n] };
            *(v2h*)&Bt[n][kk] = pk;
        }
        __syncthreads();

        v16h af = load_a_frag_rm(&As[0][0], 40, wave * 16, 0, lane);
#pragma unroll
        for (int nt = 0; nt < 4; ++nt) {
            v16h bf = load_b_frag_kt(&Bt[0][0], 40, 0, nt * 16, lane);
            acc[nt] = __builtin_amdgcn_wmma_f32_16x16x32_f16(
                false, af, false, bf, (short)0, acc[nt], false, false);
        }
    }

#pragma unroll
    for (int nt = 0; nt < 4; ++nt) {
#pragma unroll
        for (int r = 0; r < 8; ++r) {
            int row = m0 + wave * 16 + r + hid * 8;
            int col = n0 + nt * 16 + l16;
            C[(size_t)row * N + col] = (OT)(acc[nt][r] + bias[col]);
        }
    }
}

// ---------------------------------------------------------------------------
// Kernel 3: flash attention for one (b, h, 64-row i-tile), qkv in f16.
// Q fragments (A layout) and K fragments ("transposed-B" layout) are both
// contiguous along d in the qkv tensor -> loaded straight from global memory.
// Only V (j<->d transpose) and P (C->A relayout) go through LDS.
// ---------------------------------------------------------------------------
__global__ __launch_bounds__(128) void attn_kernel(
    const _Float16* __restrict__ qkv,     // (B,S,3,H,DH) f16
    const float* __restrict__ gf,         // (B,S,S,2)
    const float* __restrict__ pos_w,      // (S,S,H)
    const unsigned char* __restrict__ mask, // (B,S) bool
    const float* __restrict__ w_geo,      // (2,H)
    const float* __restrict__ b_geo,      // (H)
    _Float16* __restrict__ attn_out)      // (B,S,D) f16
{
    __shared__ _Float16 Vt[64][72];   // [d][j]
    __shared__ _Float16 Ps[64][72];   // [i][j]

    int tid = threadIdx.x;
    int wave = tid >> 5, lane = tid & 31;
    int hid = lane >> 4, l16 = lane & 15;
    int i0 = blockIdx.x * 64;
    int h  = blockIdx.y;
    int b  = blockIdx.z;

    const int ldq = 3 * kH * kDH;  // 1536
    const _Float16* qbase = qkv + ((size_t)b * kS * 3 + 0) * kH * kDH + h * kDH;
    const _Float16* kbase = qkv + ((size_t)b * kS * 3 + 1) * kH * kDH + h * kDH;
    const _Float16* vbase = qkv + ((size_t)b * kS * 3 + 2) * kH * kDH + h * kDH;

    float wg0 = w_geo[h];
    float wg1 = w_geo[kH + h];
    float bg  = b_geo[h];
    const float scale = 0.125f;  // 1/sqrt(64)

    v16h qa0 = load_a_frag_rm(qbase, ldq, i0 + wave * 16, 0, lane);
    v16h qa1 = load_a_frag_rm(qbase, ldq, i0 + wave * 16, 32, lane);

    v8f o[4] = {};
    float mrow[8], lrow[8];
#pragma unroll
    for (int r = 0; r < 8; ++r) { mrow[r] = -1e30f; lrow[r] = 0.0f; }

    for (int j0 = 0; j0 < kS; j0 += 64) {
        __syncthreads();
        for (int idx = tid; idx < 64 * 32; idx += 128) {
            int c  = idx & 63;          // d
            int r2 = (idx >> 6) * 2;    // j pair
            v2h pk = { vbase[(size_t)(j0 + r2) * ldq + c],
                       vbase[(size_t)(j0 + r2 + 1) * ldq + c] };
            *(v2h*)&Vt[c][r2] = pk;
        }
        __syncthreads();

        // S tile = Q K^T (K fragments straight from global)
        v8f sf[4];
#pragma unroll
        for (int nt = 0; nt < 4; ++nt) {
            v8f z = {};
            v16h kb0 = load_b_frag_kt(kbase + (size_t)j0 * ldq, ldq, 0, nt * 16, lane);
            z = __builtin_amdgcn_wmma_f32_16x16x32_f16(false, qa0, false, kb0, (short)0, z, false, false);
            v16h kb1 = load_b_frag_kt(kbase + (size_t)j0 * ldq, ldq, 32, nt * 16, lane);
            z = __builtin_amdgcn_wmma_f32_16x16x32_f16(false, qa1, false, kb1, (short)0, z, false, false);
            sf[nt] = z;
        }

        // scale + fused geometric bias + positional bias + mask
#pragma unroll
        for (int nt = 0; nt < 4; ++nt) {
            int gj = j0 + nt * 16 + l16;
            float mv = mask[(size_t)b * kS + gj] ? 0.0f : -1e30f;
#pragma unroll
            for (int r = 0; r < 8; ++r) {
                int gi = i0 + wave * 16 + r + hid * 8;
                size_t pij = (size_t)gi * kS + gj;          // batch-independent (i,j)
                size_t gij = (size_t)b * kS * kS + pij;     // batched (i,j)
                float g0 = gf[gij * 2 + 0];
                float g1 = gf[gij * 2 + 1];
                float pw = pos_w[pij * kH + h];
                sf[nt][r] = sf[nt][r] * scale + g0 * wg0 + g1 * wg1 + bg + pw + mv;
            }
        }

        // online softmax; row (r + hid*8) lives in 16 lanes of this half-wave
#pragma unroll
        for (int r = 0; r < 8; ++r) {
            float mx = fmaxf(fmaxf(sf[0][r], sf[1][r]), fmaxf(sf[2][r], sf[3][r]));
#pragma unroll
            for (int m = 8; m >= 1; m >>= 1) mx = fmaxf(mx, __shfl_xor(mx, m, 32));
            float newm  = fmaxf(mrow[r], mx);
            float alpha = __expf(mrow[r] - newm);
            float rs = 0.0f;
#pragma unroll
            for (int nt = 0; nt < 4; ++nt) {
                float p = __expf(sf[nt][r] - newm);
                sf[nt][r] = p;
                rs += p;
            }
#pragma unroll
            for (int m = 8; m >= 1; m >>= 1) rs += __shfl_xor(rs, m, 32);
            lrow[r] = lrow[r] * alpha + rs;
            mrow[r] = newm;
#pragma unroll
            for (int nt = 0; nt < 4; ++nt) o[nt][r] *= alpha;
        }

        // P: C-layout regs -> LDS row-major (wave-private rows)
#pragma unroll
        for (int nt = 0; nt < 4; ++nt) {
#pragma unroll
            for (int r = 0; r < 8; ++r) {
                Ps[wave * 16 + r + hid * 8][nt * 16 + l16] = (_Float16)sf[nt][r];
            }
        }
        __syncthreads();

        // O += P @ V
        v16h pa0 = load_a_frag_rm(&Ps[0][0], 72, wave * 16, 0, lane);
        v16h pa1 = load_a_frag_rm(&Ps[0][0], 72, wave * 16, 32, lane);
#pragma unroll
        for (int dt = 0; dt < 4; ++dt) {
            v16h vb0 = load_b_frag_kt(&Vt[0][0], 72, 0, dt * 16, lane);
            o[dt] = __builtin_amdgcn_wmma_f32_16x16x32_f16(false, pa0, false, vb0, (short)0, o[dt], false, false);
            v16h vb1 = load_b_frag_kt(&Vt[0][0], 72, 32, dt * 16, lane);
            o[dt] = __builtin_amdgcn_wmma_f32_16x16x32_f16(false, pa1, false, vb1, (short)0, o[dt], false, false);
        }
    }

#pragma unroll
    for (int dt = 0; dt < 4; ++dt) {
#pragma unroll
        for (int r = 0; r < 8; ++r) {
            int gi = i0 + wave * 16 + r + hid * 8;
            int gd = h * kDH + dt * 16 + l16;
            attn_out[((size_t)b * kS + gi) * kD + gd] = (_Float16)(o[dt][r] / lrow[r]);
        }
    }
}

// ---------------------------------------------------------------------------
extern "C" void kernel_launch(void* const* d_in, const int* in_sizes, int n_in,
                              void* d_out, int out_size, void* d_ws, size_t ws_size,
                              hipStream_t stream) {
    const float* x      = (const float*)d_in[0];
    const float* gfe    = (const float*)d_in[1];
    const float* pe     = (const float*)d_in[2];
    const unsigned char* mask = (const unsigned char*)d_in[3];
    const float* W_qkv  = (const float*)d_in[4];
    const float* b_qkv  = (const float*)d_in[5];
    const float* W_out  = (const float*)d_in[6];
    const float* b_out  = (const float*)d_in[7];
    const float* W_geo  = (const float*)d_in[8];
    const float* b_geo  = (const float*)d_in[9];
    const float* W_pos  = (const float*)d_in[10];
    const float* b_pos  = (const float*)d_in[11];
    float* out = (float*)d_out;

    const int M = kB * kS;  // 2048
    // Workspace: qkv f16 (6MB) @0, pos_w f32 (8MB) @8MB, attn f16 (2MB) @16MB
    _Float16* qkv   = (_Float16*)d_ws;
    float*    pos_w = (float*)((char*)d_ws + (size_t)8 * 1024 * 1024);
    _Float16* attn  = (_Float16*)((char*)d_ws + (size_t)16 * 1024 * 1024);

    // 1) positional bias GEMM (streams 512MB PE once, WMMA to stay BW-bound)
    pos_bias_wmma<<<dim3(kS * kS / 64), 128, 0, stream>>>(pe, W_pos, b_pos, pos_w);

    // 2) qkv = x @ W_qkv + b_qkv   (2048 x 1536 x 512), f16 out
    gemm_bias_wmma<float, _Float16><<<dim3(3 * kD / 64, M / 64), 128, 0, stream>>>(
        x, W_qkv, b_qkv, qkv, M, 3 * kD, kD);

    // 3) flash attention with fused geo bias + pos bias, f16 out
    attn_kernel<<<dim3(kS / 64, kH, kB), 128, 0, stream>>>(
        qkv, gfe, pos_w, mask, W_geo, b_geo, attn);

    // 4) out = attn @ W_out + b_out (2048 x 512 x 512), f32 out
    gemm_bias_wmma<_Float16, float><<<dim3(kD / 64, M / 64), 128, 0, stream>>>(
        attn, W_out, b_out, out, M, kD, kD);
}
